// AMLSTM_58523224375868
// MI455X (gfx1250) — compile-verified, matrix operand order checked by reading, not run on hardware
//
#include <hip/hip_runtime.h>
#include <hip/hip_bf16.h>
#include <math.h>

// ---------------------------------------------------------------------------
// Problem constants
// ---------------------------------------------------------------------------
constexpr int Bc = 64, Sc = 2048, Hc = 1024, Ec = 512, Ac = 512, Mc = 120;
constexpr float EPSV = 1e-40f;

typedef __attribute__((ext_vector_type(16))) __bf16 v16bf;
typedef __attribute__((ext_vector_type(8)))  __bf16 v8bf;
typedef __attribute__((ext_vector_type(8)))  float  v8f;

#define LSTR 40   // LDS row stride in bf16 elements (16B aligned, bank-spread)

// ---------------------------------------------------------------------------
// CDNA5 async global->LDS copy (16B per lane), tracked by ASYNCcnt.
// vdst VGPR = LDS byte offset (addr[31:0] of the generic pointer), GV mode.
// ---------------------------------------------------------------------------
__device__ __forceinline__ void async_copy16(const void* gsrc, const void* ldst) {
    unsigned loff = (unsigned)(uintptr_t)ldst;          // LDS offset = low 32 bits
    unsigned long long ga = (unsigned long long)(uintptr_t)gsrc;
    asm volatile("global_load_async_to_lds_b128 %0, %1, off"
                 :: "v"(loff), "v"(ga) : "memory");
}
__device__ __forceinline__ void async_wait0() {
    asm volatile("s_wait_asynccnt 0x0" ::: "memory");
}

// ---------------------------------------------------------------------------
// WMMA fragment load from an LDS tile (row-major, stride LSTR bf16).
// 16-bit A/B layout (ISA 7.12.2): lane L -> row/col (L&15), kbase = (L<16)?0:8
// elements 0..7  = k in [kbase, kbase+8)
// elements 8..15 = k in [kbase+16, kbase+24)
// ---------------------------------------------------------------------------
__device__ __forceinline__ v16bf load_frag(const __bf16* base, int lane) {
    int r  = lane & 15;
    int kb = (lane >> 4) << 3;                    // 0 or 8
    const __bf16* p = base + r * LSTR + kb;
    v8bf lo = *(const v8bf*)(p);
    v8bf hi = *(const v8bf*)(p + 16);
    return __builtin_shufflevector(lo, hi, 0,1,2,3,4,5,6,7,8,9,10,11,12,13,14,15);
}

// ---------------------------------------------------------------------------
// f32 -> bf16 conversion (weights -> workspace, once per call)
// ---------------------------------------------------------------------------
__global__ void cvt_bf16_kernel(const float* __restrict__ src,
                                __bf16* __restrict__ dst, int n) {
    int i = blockIdx.x * blockDim.x + threadIdx.x;
    if (i < n) dst[i] = (__bf16)src[i];
}

// ---------------------------------------------------------------------------
// Generic tiny GEMV: out[b,n] = act( dot(X[b,:K], W[n,:K]) + bias[n] )
// act: 0 = identity, 1 = tanh
// ---------------------------------------------------------------------------
__global__ void gemv_kernel(const float* __restrict__ X, const float* __restrict__ W,
                            const float* __restrict__ bias, float* __restrict__ out,
                            int K, int N, int act) {
    int id = blockIdx.x * blockDim.x + threadIdx.x;
    int b = id / N, n = id % N;
    const float4* x = (const float4*)(X + (size_t)b * K);
    const float4* w = (const float4*)(W + (size_t)n * K);
    float acc = 0.f;
    for (int k = 0; k < K / 4; ++k) {
        float4 xa = x[k], wa = w[k];
        acc += xa.x * wa.x + xa.y * wa.y + xa.z * wa.z + xa.w * wa.w;
    }
    acc += bias[n];
    if (act) acc = tanhf(acc);
    out[id] = acc;
}

// ---------------------------------------------------------------------------
// NTM soft read: score->softmax(M)->weighted read. One block per batch b.
// ---------------------------------------------------------------------------
__global__ void ntm_read_kernel(const float* __restrict__ vvec,
                                const float* __restrict__ mem,
                                float* __restrict__ mt) {
    __shared__ float v_s[Ec];
    __shared__ float sc_s[128];
    __shared__ float inv_s;
    int b = blockIdx.x, t = threadIdx.x;
    v_s[t]       = vvec[b * Ec + t];
    v_s[t + 256] = vvec[b * Ec + t + 256];
    __syncthreads();
    if (t < Mc) {
        const float* row = mem + ((size_t)b * Mc + t) * Ec;
        float acc = 0.f;
        for (int e = 0; e < Ec; ++e) acc += v_s[e] * row[e];
        sc_s[t] = acc;
    }
    __syncthreads();
    if (t == 0) {
        float mx = -INFINITY;
        for (int m = 0; m < Mc; ++m) mx = fmaxf(mx, sc_s[m]);
        float sum = 0.f;
        for (int m = 0; m < Mc; ++m) { float e = __expf(sc_s[m] - mx); sc_s[m] = e; sum += e; }
        inv_s = 1.f / sum;
    }
    __syncthreads();
    float inv = inv_s;
    for (int e = t; e < Ec; e += 256) {
        float acc = 0.f;
        for (int m = 0; m < Mc; ++m)
            acc += sc_s[m] * mem[((size_t)b * Mc + m) * Ec + e];
        mt[b * Ec + e] = acc * inv;
    }
}

// ---------------------------------------------------------------------------
// gate[b,m] = sigmoid(dot(mem[b,m,:], inputs[b,:]))   one wave per row
// ---------------------------------------------------------------------------
__global__ void gate_kernel(const float* __restrict__ mem,
                            const float* __restrict__ inp,
                            float* __restrict__ gate) {
    int wid  = (blockIdx.x * blockDim.x + threadIdx.x) >> 5;
    int lane = threadIdx.x & 31;
    if (wid >= Bc * Mc) return;
    int b = wid / Mc;
    const float* row = mem + (size_t)wid * Ec;
    const float* x   = inp + b * Ec;
    float acc = 0.f;
    for (int e = lane; e < Ec; e += 32) acc += row[e] * x[e];
    #pragma unroll
    for (int off = 16; off > 0; off >>= 1) acc += __shfl_xor(acc, off, 32);
    if (lane == 0) gate[wid] = 1.f / (1.f + __expf(-acc));
}

// ---------------------------------------------------------------------------
// WMMA: candidate GEMM  acc = mem @ W_mem^T  (rows = B*M = 7680, N = 512, K = 512)
// fused epilogue: cand = acc + inC[b,:] + b_mem ; new = (1-g)*mem + g*cand ;
// inf/nan -> EPS ; written straight into d_out new_mem region.
// Block = 256 thr (8 waves), 16 rows x 512 cols; wave w -> cols [w*64, w*64+64).
// ---------------------------------------------------------------------------
__global__ __launch_bounds__(256)
void memwrite_wmma_kernel(const float* __restrict__ mem, const __bf16* __restrict__ Wm,
                          const float* __restrict__ inC, const float* __restrict__ b_mem,
                          const float* __restrict__ gate, float* __restrict__ newmem) {
    __shared__ __attribute__((aligned(16))) __bf16 Als[16 * LSTR];
    __shared__ __attribute__((aligned(16))) __bf16 Bls[Ec * LSTR];
    int t = threadIdx.x, wv = t >> 5, lane = t & 31;
    int rowbase = blockIdx.x * 16;

    v8f acc[4];
    v8f zf = {0.f,0.f,0.f,0.f,0.f,0.f,0.f,0.f};
    #pragma unroll
    for (int i = 0; i < 4; ++i) acc[i] = zf;

    for (int kc = 0; kc < Ec; kc += 32) {
        // stage A: 16 rows x 32 k, f32 -> bf16 (conversion in flight -> VALU path)
        for (int i = t; i < 16 * 32; i += 256) {
            int r = i >> 5, kk = i & 31;
            Als[r * LSTR + kk] = (__bf16)mem[((size_t)(rowbase + r)) * Ec + kc + kk];
        }
        // stage B: 512 rows x 32 k of bf16 weights -> async DMA into LDS
        for (int i = t; i < Ec * 2; i += 256) {
            int n = i >> 1, c = (i & 1) * 16;
            async_copy16(&Wm[(size_t)n * Ec + kc + c],     &Bls[n * LSTR + c]);
            async_copy16(&Wm[(size_t)n * Ec + kc + c + 8], &Bls[n * LSTR + c + 8]);
        }
        async_wait0();
        __syncthreads();
        v16bf afrag = load_frag(Als, lane);
        #pragma unroll
        for (int ct = 0; ct < 4; ++ct) {
            int n0 = wv * 64 + ct * 16;
            v16bf bfrag = load_frag(Bls + n0 * LSTR, lane);
            acc[ct] = __builtin_amdgcn_wmma_f32_16x16x32_bf16(
                false, afrag, false, bfrag, (short)0, acc[ct], false, false);
        }
        __syncthreads();
    }

    // epilogue: D layout -> lane: N = lane&15, M = r + 8*(lane>>4)
    int nl = lane & 15, ml = (lane >> 4) << 3;
    int b0 = rowbase / Mc;             // uniform: one scalar division per block
    int r0 = rowbase - b0 * Mc;        // tile crosses at most one batch boundary
    float g[8];
    #pragma unroll
    for (int r = 0; r < 8; ++r) g[r] = gate[rowbase + ml + r];
    #pragma unroll
    for (int ct = 0; ct < 4; ++ct) {
        int n = wv * 64 + ct * 16 + nl;
        float bm = b_mem[n];
        #pragma unroll
        for (int r = 0; r < 8; ++r) {
            int d = ml + r;
            int b = b0 + ((r0 + d) >= Mc ? 1 : 0);
            int grow = rowbase + d;
            float cand = acc[ct][r] + inC[b * Ec + n] + bm;
            float old  = mem[(size_t)grow * Ec + n];
            float val  = (1.f - g[r]) * old + g[r] * cand;
            if (isinf(val) || isnan(val)) val = EPSV;
            newmem[(size_t)grow * Ec + n] = val;
        }
    }
}

// ---------------------------------------------------------------------------
// WMMA: big fused attention-score kernel.
// acc = enc @ W_enc^T   (rows = B*S = 131072, N = A = 512, K = H = 1024)
// epilogue: score[row] = sum_a attn_v[a] * tanh(acc + b_enc[a] + qp[b,a])
// Block = 256 thr (8 waves), 32 rows x 512 cols.
// wave w: row-tile (w&1) (16 rows), cols [(w>>1)*128, +128) = 8 WMMA tiles.
// ---------------------------------------------------------------------------
__global__ __launch_bounds__(256)
void encscore_wmma_kernel(const float* __restrict__ enc, const __bf16* __restrict__ We,
                          const float* __restrict__ qp, const float* __restrict__ b_enc,
                          const float* __restrict__ attn_v, float* __restrict__ score) {
    __shared__ __attribute__((aligned(16))) __bf16 Als[32 * LSTR];
    __shared__ __attribute__((aligned(16))) __bf16 Bls[Ac * LSTR];
    __shared__ float qpb_s[Ac];
    __shared__ float av_s[Ac];
    __shared__ float sc_s[32];

    int t = threadIdx.x, wv = t >> 5, lane = t & 31;
    size_t rowbase = (size_t)blockIdx.x * 32;     // flattened b*S + s
    int b = (int)(rowbase / Sc);                  // S % 32 == 0 -> constant per block

    for (int i = t; i < Ac; i += 256) {
        qpb_s[i] = qp[b * Ac + i] + b_enc[i];
        av_s[i]  = attn_v[i];
    }
    if (t < 32) sc_s[t] = 0.f;

    v8f acc[8];
    v8f zf = {0.f,0.f,0.f,0.f,0.f,0.f,0.f,0.f};
    #pragma unroll
    for (int i = 0; i < 8; ++i) acc[i] = zf;

    for (int kc = 0; kc < Hc; kc += 32) {
        // stage A: 32 rows x 32 k of encoder, f32 -> bf16 (VALU path)
        for (int i = t; i < 32 * 32; i += 256) {
            int r = i >> 5, kk = i & 31;
            Als[r * LSTR + kk] = (__bf16)enc[(rowbase + r) * Hc + kc + kk];
        }
        // stage B: 512 W_enc rows x 32 k bf16 -> async DMA into LDS
        for (int i = t; i < Ac * 2; i += 256) {
            int n = i >> 1, c = (i & 1) * 16;
            async_copy16(&We[(size_t)n * Hc + kc + c],     &Bls[n * LSTR + c]);
            async_copy16(&We[(size_t)n * Hc + kc + c + 8], &Bls[n * LSTR + c + 8]);
        }
        async_wait0();
        __syncthreads();
        v16bf afrag = load_frag(Als + (wv & 1) * 16 * LSTR, lane);
        #pragma unroll
        for (int ct = 0; ct < 8; ++ct) {
            int n0 = (wv >> 1) * 128 + ct * 16;
            v16bf bfrag = load_frag(Bls + n0 * LSTR, lane);
            acc[ct] = __builtin_amdgcn_wmma_f32_16x16x32_bf16(
                false, afrag, false, bfrag, (short)0, acc[ct], false, false);
        }
        __syncthreads();
    }

    // epilogue: tanh + dot-with-attn_v, reduce per row via LDS atomics
    int nl = lane & 15, ml = (lane >> 4) << 3;
    #pragma unroll
    for (int r = 0; r < 8; ++r) {
        float part = 0.f;
        #pragma unroll
        for (int ct = 0; ct < 8; ++ct) {
            int n = (wv >> 1) * 128 + ct * 16 + nl;
            part += av_s[n] * tanhf(acc[ct][r] + qpb_s[n]);
        }
        atomicAdd(&sc_s[(wv & 1) * 16 + ml + r], part);
    }
    __syncthreads();
    if (t < 32) score[rowbase + t] = sc_s[t];
}

// ---------------------------------------------------------------------------
// Softmax over S per batch, in place. One block per b.
// ---------------------------------------------------------------------------
__global__ void softmax_kernel(float* __restrict__ score) {
    __shared__ float red_s[256];
    int b = blockIdx.x, t = threadIdx.x;
    float* row = score + (size_t)b * Sc;
    float mx = -INFINITY;
    for (int s = t; s < Sc; s += 256) mx = fmaxf(mx, row[s]);
    red_s[t] = mx; __syncthreads();
    for (int o = 128; o > 0; o >>= 1) { if (t < o) red_s[t] = fmaxf(red_s[t], red_s[t + o]); __syncthreads(); }
    mx = red_s[0]; __syncthreads();
    float sum = 0.f;
    for (int s = t; s < Sc; s += 256) { float e = __expf(row[s] - mx); row[s] = e; sum += e; }
    red_s[t] = sum; __syncthreads();
    for (int o = 128; o > 0; o >>= 1) { if (t < o) red_s[t] += red_s[t + o]; __syncthreads(); }
    float inv = 1.f / red_s[0];
    for (int s = t; s < Sc; s += 256) row[s] *= inv;
}

// ---------------------------------------------------------------------------
// attn_value[b,h] = sum_s sim2[b,s] * enc[b,s,h]  (HBM-bound, coalesced over h)
// Grid: B*4 blocks of 256 (4 h-slices per b).
// ---------------------------------------------------------------------------
__global__ void attn_reduce_kernel(const float* __restrict__ enc,
                                   const float* __restrict__ sim,
                                   float* __restrict__ out) {
    int b = blockIdx.x >> 2;
    int h = (blockIdx.x & 3) * 256 + threadIdx.x;
    const float* e = enc + (size_t)b * Sc * Hc + h;
    const float* w = sim + (size_t)b * Sc;
    float acc = 0.f;
    for (int s = 0; s < Sc; ++s) {
        if (s + 16 < Sc) __builtin_prefetch(&e[(size_t)(s + 16) * Hc], 0, 1);
        acc += w[s] * e[(size_t)s * Hc];
    }
    out[b * Hc + h] = acc;
}

// ---------------------------------------------------------------------------
// lstm_inp = concat([inputs, attn_value, mt], -1)   [B, 2048]
// ---------------------------------------------------------------------------
__global__ void concat_kernel(const float* __restrict__ inp, const float* __restrict__ attn,
                              const float* __restrict__ mt, float* __restrict__ out) {
    int id = blockIdx.x * 256 + threadIdx.x;     // B * 2048
    int b = id >> 11, j = id & 2047;
    float v;
    if (j < Ec)            v = inp[b * Ec + j];
    else if (j < Ec + Hc)  v = attn[b * Hc + (j - Ec)];
    else                   v = mt[b * Ec + (j - Ec - Hc)];
    out[id] = v;
}

// ---------------------------------------------------------------------------
// Launch
// ---------------------------------------------------------------------------
extern "C" void kernel_launch(void* const* d_in, const int* in_sizes, int n_in,
                              void* d_out, int out_size, void* d_ws, size_t ws_size,
                              hipStream_t stream) {
    (void)in_sizes; (void)n_in; (void)out_size; (void)ws_size;
    const float* enc   = (const float*)d_in[0];
    const float* dech  = (const float*)d_in[1];
    const float* decc  = (const float*)d_in[2];
    const float* inp   = (const float*)d_in[3];
    const float* mem   = (const float*)d_in[4];
    const float* W_h1  = (const float*)d_in[5];
    const float* b_h1  = (const float*)d_in[6];
    const float* W_h2  = (const float*)d_in[7];
    const float* b_h2  = (const float*)d_in[8];
    const float* W_in1 = (const float*)d_in[9];
    const float* b_in1 = (const float*)d_in[10];
    const float* W_mem = (const float*)d_in[11];
    const float* b_mem = (const float*)d_in[12];
    const float* W_enc = (const float*)d_in[13];
    const float* b_enc = (const float*)d_in[14];
    const float* attnv = (const float*)d_in[15];

    float* out    = (float*)d_out;
    float* newmem = out + (size_t)Bc * 2048;     // tuple part 2

    // workspace layout (floats, then 256B-aligned bf16 region)
    float* ws      = (float*)d_ws;
    float* v_ws    = ws;                                   // B*E
    float* inC_ws  = v_ws   + (size_t)Bc * Ec;             // B*E
    float* qp_ws   = inC_ws + (size_t)Bc * Ec;             // B*A
    float* mt_ws   = qp_ws  + (size_t)Bc * Ac;             // B*E
    float* gate_ws = mt_ws  + (size_t)Bc * Ec;             // B*M (pad 8192)
    float* sc_ws   = gate_ws + 8192;                       // B*S
    float* attn_ws = sc_ws  + (size_t)Bc * Sc;             // B*H
    size_t fend_bytes = ((char*)(attn_ws + (size_t)Bc * Hc) - (char*)d_ws);
    size_t bf_off = (fend_bytes + 255) & ~(size_t)255;
    __bf16* WencB = (__bf16*)((char*)d_ws + bf_off);       // A*H bf16
    __bf16* WmemB = WencB + (size_t)Ac * Hc;               // E*E bf16

    // 1. weight conversions
    cvt_bf16_kernel<<<(Ac * Hc + 255) / 256, 256, 0, stream>>>(W_enc, WencB, Ac * Hc);
    cvt_bf16_kernel<<<(Ec * Ec + 255) / 256, 256, 0, stream>>>(W_mem, WmemB, Ec * Ec);

    // 2. small GEMVs
    gemv_kernel<<<(Bc * Ec) / 256, 256, 0, stream>>>(dech, W_h1, b_h1, v_ws,   Hc, Ec, 1);
    gemv_kernel<<<(Bc * Ec) / 256, 256, 0, stream>>>(inp,  W_in1, b_in1, inC_ws, Ec, Ec, 0);
    gemv_kernel<<<(Bc * Ac) / 256, 256, 0, stream>>>(decc, W_h2, b_h2, qp_ws,  Hc, Ac, 0);

    // 3. NTM soft read -> mt
    ntm_read_kernel<<<Bc, 256, 0, stream>>>(v_ws, mem, mt_ws);

    // 4. write gates
    gate_kernel<<<(Bc * Mc * 32) / 256, 256, 0, stream>>>(mem, inp, gate_ws);

    // 5. candidate GEMM + gated blend -> new_mem (WMMA + async LDS staging)
    memwrite_wmma_kernel<<<(Bc * Mc) / 16, 256, 0, stream>>>(
        mem, WmemB, inC_ws, b_mem, gate_ws, newmem);

    // 6. big fused enc @ W_enc^T -> tanh -> dot attn_v (WMMA + async LDS staging)
    encscore_wmma_kernel<<<(Bc * Sc) / 32, 256, 0, stream>>>(
        enc, WencB, qp_ws, b_enc, attnv, sc_ws);

    // 7. softmax over S (in place)
    softmax_kernel<<<Bc, 256, 0, stream>>>(sc_ws);

    // 8. weighted encoder reduction -> attn_value
    attn_reduce_kernel<<<Bc * 4, 256, 0, stream>>>(enc, sc_ws, attn_ws);

    // 9. concat -> lstm_inp
    concat_kernel<<<(Bc * 2048) / 256, 256, 0, stream>>>(inp, attn_ws, mt_ws, out);
}
